// MLA_7988639170878
// MI455X (gfx1250) — matrix operand order repaired and checked by reading, not compiled, not run on hardware
//
#include <hip/hip_runtime.h>
#include <cmath>

// ---------- types ----------
typedef __attribute__((ext_vector_type(16))) __bf16        v16bf;
typedef __attribute__((ext_vector_type(8)))  float         v8f;
typedef __attribute__((ext_vector_type(4)))  unsigned int  u32x4;

union BF16x16 { u32x4 q[2]; v16bf v; };

__device__ __forceinline__ unsigned short f2bf(float f) {
    unsigned u = __builtin_bit_cast(unsigned, f);
    unsigned r = u + 0x7fffu + ((u >> 16) & 1u);
    return (unsigned short)(r >> 16);
}
__device__ __forceinline__ float bf2f(unsigned short u) {
    return __builtin_bit_cast(float, ((unsigned)u) << 16);
}

// ---------- CDNA5 async memory->LDS copy (32 bytes), ASYNCcnt tracked ----------
__device__ __forceinline__ void async_copy32(void* lds, const void* gmem) {
    unsigned l = (unsigned)(unsigned long long)lds;   // LDS aperture: low 32 bits = LDS offset
    asm volatile("global_load_async_to_lds_b128 %0, %1, off\n\t"
                 "global_load_async_to_lds_b128 %0, %1, off offset:16"
                 :: "v"(l), "v"(gmem) : "memory");
}
__device__ __forceinline__ void wait_async0() {
    asm volatile("s_wait_asynccnt 0x0" ::: "memory");
}

#define S_LEN 2048
#define HN    16

// ---------- elementwise f32 -> bf16 ----------
__global__ void cvt_f32_bf16(const float* __restrict__ in, unsigned short* __restrict__ out, long n) {
    long i = (long)blockIdx.x * blockDim.x + threadIdx.x;
    if (i < n) out[i] = f2bf(in[i]);
}

// ---------- generic u16 transpose: out[z][c][r] = in[z][r][c] ----------
__global__ void transpose_u16(const unsigned short* __restrict__ in, unsigned short* __restrict__ out,
                              int R, int C, long inZ, long outZ) {
    long z = blockIdx.z;
    long i = (long)blockIdx.x * blockDim.x + threadIdx.x;
    if (i >= (long)R * C) return;
    int r = (int)(i / C), c = (int)(i % C);
    out[z * outZ + (long)c * R + r] = in[z * inZ + i];
}

// ---------- tiled NT GEMM, bf16 in, fp32 acc, WMMA 16x16x32 ----------
// C[M,N] = A[M,K] * B[N,K]^T ; M % 128 == 0, K % 32 == 0, N guarded.
// Double-buffered LDS staged with GLOBAL_LOAD_ASYNC_TO_LDS_B128.
__global__ __launch_bounds__(256) void gemm_nt_bf16(
    const unsigned short* __restrict__ A, const unsigned short* __restrict__ B,
    float* __restrict__ Cf, unsigned short* __restrict__ Cb,
    int M, int N, int K, int lda, int ldb, int ldc,
    long sA, long sB, long sC)
{
    __shared__ unsigned short As[2][128][48];
    __shared__ unsigned short Bs[2][128][48];

    const int z = blockIdx.z;
    A += (long)z * sA;  B += (long)z * sB;
    const long coff = (long)z * sC;
    const int m0 = blockIdx.y * 128, n0 = blockIdx.x * 128;
    const int tid = threadIdx.x, lane = tid & 31, wv = tid >> 5;
    const int wm = (wv & 3) * 32, wn = (wv >> 2) * 64;
    const int lh = lane & 15, lq = lane >> 4;

    v8f acc[2][4];
#pragma unroll
    for (int i = 0; i < 2; ++i)
#pragma unroll
        for (int jx = 0; jx < 4; ++jx)
            acc[i][jx] = (v8f){0.f,0.f,0.f,0.f,0.f,0.f,0.f,0.f};

    const int lr = tid >> 1;        // 0..127 : tile row
    const int lk = (tid & 1) * 16;  // 0 or 16
    const int bn = n0 + lr;

    // stage one 128x32 A tile + B tile into buffer `buf` for k-offset k0
    auto stage = [&](int buf, int k0) {
        async_copy32(&As[buf][lr][lk], A + (long)(m0 + lr) * lda + k0 + lk);
        if (bn < N) {
            async_copy32(&Bs[buf][lr][lk], B + (long)bn * ldb + k0 + lk);
        } else {
            u32x4 zz = (u32x4){0u,0u,0u,0u};
            *(u32x4*)&Bs[buf][lr][lk]     = zz;
            *(u32x4*)&Bs[buf][lr][lk + 8] = zz;
        }
    };

    stage(0, 0);
    const int nk = K >> 5;
    for (int it = 0; it < nk; ++it) {
        wait_async0();          // our async stores into LDS for buffer `cur` done
        __syncthreads();        // everyone's done (barrier flushes dscnt for zero-fills)
        const int cur = it & 1;
        if (it + 1 < nk) stage(cur ^ 1, (it + 1) << 5);

        BF16x16 af[2], bfr[4];
#pragma unroll
        for (int mi = 0; mi < 2; ++mi) {
            af[mi].q[0] = *(const u32x4*)&As[cur][wm + mi*16 + lh][(lq << 3)];
            af[mi].q[1] = *(const u32x4*)&As[cur][wm + mi*16 + lh][16 + (lq << 3)];
        }
#pragma unroll
        for (int ni = 0; ni < 4; ++ni) {
            bfr[ni].q[0] = *(const u32x4*)&Bs[cur][wn + ni*16 + lh][(lq << 3)];
            bfr[ni].q[1] = *(const u32x4*)&Bs[cur][wn + ni*16 + lh][16 + (lq << 3)];
        }
#pragma unroll
        for (int mi = 0; mi < 2; ++mi)
#pragma unroll
            for (int ni = 0; ni < 4; ++ni)
                acc[mi][ni] = __builtin_amdgcn_wmma_f32_16x16x32_bf16(
                    false, af[mi].v, false, bfr[ni].v, (short)0, acc[mi][ni], false, false);
    }

    // epilogue
#pragma unroll
    for (int mi = 0; mi < 2; ++mi) {
#pragma unroll
        for (int ni = 0; ni < 4; ++ni) {
            int col = n0 + wn + ni * 16 + lh;
            if (col < N) {
#pragma unroll
                for (int j = 0; j < 8; ++j) {
                    long row = (long)m0 + wm + mi * 16 + j + (lq << 3);
                    long idx = coff + row * ldc + col;
                    float v = acc[mi][ni][j];
                    if (Cb) Cb[idx] = f2bf(v); else Cf[idx] = v;
                }
            }
        }
    }
}

// ---------- split q into nope/pe, apply RoPE to pe ----------
__global__ void rope_q(const unsigned short* __restrict__ qraw, const float* __restrict__ freqs,
                       unsigned short* __restrict__ qn, unsigned short* __restrict__ qp) {
    int bs = blockIdx.x;            // b*S + s
    int s  = bs & (S_LEN - 1);
    for (int i = threadIdx.x; i < 2048; i += 256) {      // H*128
        int h = i >> 7, d = i & 127;
        qn[(long)bs * 2048 + i] = qraw[(long)bs * 3072 + h * 192 + d];
    }
    for (int i = threadIdx.x; i < 512; i += 256) {       // H*32 pairs
        int h = i >> 5, fi = i & 31;
        float x0 = bf2f(qraw[(long)bs * 3072 + h * 192 + 128 + 2 * fi]);
        float x1 = bf2f(qraw[(long)bs * 3072 + h * 192 + 128 + 2 * fi + 1]);
        float sn, cn;
        __sincosf(freqs[s * 32 + fi], &sn, &cn);
        qp[(long)bs * 1024 + h * 64 + 2 * fi]     = f2bf(x0 * cn - x1 * sn);
        qp[(long)bs * 1024 + h * 64 + 2 * fi + 1] = f2bf(x0 * sn + x1 * cn);
    }
}

// ---------- RMSNorm kv -> bf16; RoPE k_pe -> bf16 ----------
__global__ void kv_prep(const unsigned short* __restrict__ kvf, const float* __restrict__ w,
                        const float* __restrict__ freqs,
                        unsigned short* __restrict__ kvb, unsigned short* __restrict__ kpeb) {
    __shared__ float red[256];
    int bs = blockIdx.x;
    int s  = bs & (S_LEN - 1);
    int tid = threadIdx.x;
    float v0 = bf2f(kvf[(long)bs * 576 + tid]);
    float v1 = bf2f(kvf[(long)bs * 576 + 256 + tid]);
    red[tid] = v0 * v0 + v1 * v1;
    __syncthreads();
    for (int o = 128; o > 0; o >>= 1) {
        if (tid < o) red[tid] += red[tid + o];
        __syncthreads();
    }
    float rinv = rsqrtf(red[0] / 512.f + 1e-6f);
    kvb[(long)bs * 512 + tid]       = f2bf(v0 * rinv * w[tid]);
    kvb[(long)bs * 512 + 256 + tid] = f2bf(v1 * rinv * w[tid + 256]);
    if (tid < 32) {
        float x0 = bf2f(kvf[(long)bs * 576 + 512 + 2 * tid]);
        float x1 = bf2f(kvf[(long)bs * 576 + 512 + 2 * tid + 1]);
        float sn, cn;
        __sincosf(freqs[s * 32 + tid], &sn, &cn);
        kpeb[(long)bs * 64 + 2 * tid]     = f2bf(x0 * cn - x1 * sn);
        kpeb[(long)bs * 64 + 2 * tid + 1] = f2bf(x0 * sn + x1 * cn);
    }
}

// ---------- flash attention: 1 wave / 16 queries, 64-key blocks ----------
__global__ __launch_bounds__(32) void mla_attn(
    const unsigned short* __restrict__ qabs,  // (B*S, H, 512)
    const unsigned short* __restrict__ qpe,   // (B*S, H, 64)
    const unsigned short* __restrict__ kvb,   // (B, S, 512)
    const unsigned short* __restrict__ kpeb,  // (B, S, 64)
    const unsigned short* __restrict__ vt,    // (B, 512, S)
    unsigned short* __restrict__ attn,        // (B*S, H, 512)
    float scale)
{
    __shared__ float outacc[16 * 512];        // fp32 output accumulator (32 KB)
    __shared__ unsigned short pbuf[16 * 72];  // P tile, C-layout -> A-layout staging
    __shared__ float fac[16];

    const int lane = threadIdx.x;
    const int q0 = blockIdx.x * 16;
    const int h  = blockIdx.y;
    const int b  = blockIdx.z;
    const int lh = lane & 15, lq = lane >> 4;

    for (int i = lane; i < 16 * 512; i += 32) outacc[i] = 0.f;
    __syncthreads();

    // Q' fragments (576 = 18 * 32) held in VGPRs
    BF16x16 qa[18];
    const int qrow = q0 + lh;
    const unsigned short* qaP = qabs + ((long)b * S_LEN + qrow) * (HN * 512) + h * 512;
    const unsigned short* qpP = qpe  + ((long)b * S_LEN + qrow) * (HN * 64)  + h * 64;
#pragma unroll
    for (int kk = 0; kk < 18; ++kk) {
        const unsigned short* src = (kk < 16) ? (qaP + kk * 32) : (qpP + (kk - 16) * 32);
        qa[kk].q[0] = *(const u32x4*)(src + (lq << 3));
        qa[kk].q[1] = *(const u32x4*)(src + 16 + (lq << 3));
    }

    float m[8], l[8];
#pragma unroll
    for (int j = 0; j < 8; ++j) { m[j] = -1e30f; l[j] = 0.f; }

    const int kb_last = q0 >> 6;
    for (int kb = 0; kb <= kb_last; ++kb) {
        // prefetch next key block rows into cache (global_prefetch_b8)
        if (kb < kb_last) {
            __builtin_prefetch(kvb + ((long)b * S_LEN + (kb + 1) * 64 + lane) * 512, 0, 1);
            __builtin_prefetch(kvb + ((long)b * S_LEN + (kb + 1) * 64 + 32 + lane) * 512, 0, 1);
        }
        // ---- scores: Q'(16x576) x K'(64x576)^T ----
        v8f sc[4];
#pragma unroll
        for (int nt = 0; nt < 4; ++nt) {
            v8f a = (v8f){0.f,0.f,0.f,0.f,0.f,0.f,0.f,0.f};
            const int key = kb * 64 + nt * 16 + lh;
            const unsigned short* kP  = kvb  + ((long)b * S_LEN + key) * 512;
            const unsigned short* kpP = kpeb + ((long)b * S_LEN + key) * 64;
#pragma unroll
            for (int kk = 0; kk < 18; ++kk) {
                const unsigned short* src = (kk < 16) ? (kP + kk * 32 + (lq << 3))
                                                      : (kpP + (kk - 16) * 32 + (lq << 3));
                BF16x16 bf;
                bf.q[0] = *(const u32x4*)(src);
                bf.q[1] = *(const u32x4*)(src + 16);
                a = __builtin_amdgcn_wmma_f32_16x16x32_bf16(
                        false, qa[kk].v, false, bf.v, (short)0, a, false, false);
            }
            sc[nt] = a;
        }
        // ---- scale + causal mask ----
#pragma unroll
        for (int nt = 0; nt < 4; ++nt)
#pragma unroll
            for (int j = 0; j < 8; ++j) {
                int row = q0 + j + (lq << 3);
                int col = kb * 64 + nt * 16 + lh;
                float v = sc[nt][j] * scale;
                sc[nt][j] = (col <= row) ? v : -1e30f;
            }
        // ---- online softmax stats ----
        float mnew[8], f[8];
#pragma unroll
        for (int j = 0; j < 8; ++j) {
            float v = fmaxf(fmaxf(sc[0][j], sc[1][j]), fmaxf(sc[2][j], sc[3][j]));
#pragma unroll
            for (int o = 1; o < 16; o <<= 1) v = fmaxf(v, __shfl_xor(v, o, 32));
            mnew[j] = fmaxf(m[j], v);
            f[j] = __expf(m[j] - mnew[j]);
        }
#pragma unroll
        for (int j = 0; j < 8; ++j) {
            float sum = 0.f;
#pragma unroll
            for (int nt = 0; nt < 4; ++nt) {
                float p = __expf(sc[nt][j] - mnew[j]);
                sc[nt][j] = p;
                sum += p;
            }
#pragma unroll
            for (int o = 1; o < 16; o <<= 1) sum += __shfl_xor(sum, o, 32);
            l[j] = l[j] * f[j] + sum;
            m[j] = mnew[j];
        }
        // ---- P to LDS (C-layout write), factors to LDS ----
#pragma unroll
        for (int nt = 0; nt < 4; ++nt)
#pragma unroll
            for (int j = 0; j < 8; ++j)
                pbuf[(j + (lq << 3)) * 72 + nt * 16 + lh] = f2bf(sc[nt][j]);
        if (lh == 0) {
#pragma unroll
            for (int j = 0; j < 8; ++j) fac[(lq << 3) + j] = f[j];
        }
        __syncthreads();
        // ---- rescale accumulator ----
#pragma unroll
        for (int r = 0; r < 16; ++r) {
            float fr = fac[r];
            for (int c = lane; c < 512; c += 32) outacc[r * 512 + c] *= fr;
        }
        __syncthreads();
        // ---- P(16x64) x V(64x512) accumulate into LDS ----
        BF16x16 pa[2];
#pragma unroll
        for (int kf = 0; kf < 2; ++kf) {
            pa[kf].q[0] = *(const u32x4*)&pbuf[lh * 72 + kf * 32 + (lq << 3)];
            pa[kf].q[1] = *(const u32x4*)&pbuf[lh * 72 + kf * 32 + 16 + (lq << 3)];
        }
        const unsigned short* vbase = vt + (long)b * 512 * S_LEN + (long)lh * S_LEN
                                         + kb * 64 + (lq << 3);
#pragma unroll 4
        for (int ct = 0; ct < 32; ++ct) {
            v8f c;
#pragma unroll
            for (int j = 0; j < 8; ++j) c[j] = outacc[(j + (lq << 3)) * 512 + ct * 16 + lh];
#pragma unroll
            for (int kf = 0; kf < 2; ++kf) {
                BF16x16 bv;
                const unsigned short* src = vbase + (long)ct * 16 * S_LEN + kf * 32;
                bv.q[0] = *(const u32x4*)(src);
                bv.q[1] = *(const u32x4*)(src + 16);
                c = __builtin_amdgcn_wmma_f32_16x16x32_bf16(
                        false, pa[kf].v, false, bv.v, (short)0, c, false, false);
            }
#pragma unroll
            for (int j = 0; j < 8; ++j) outacc[(j + (lq << 3)) * 512 + ct * 16 + lh] = c[j];
        }
        __syncthreads();
    }

    // ---- epilogue: divide by l, write bf16 ----
    if (lh == 0) {
#pragma unroll
        for (int j = 0; j < 8; ++j) fac[(lq << 3) + j] = l[j];
    }
    __syncthreads();
    unsigned short* op = attn + ((long)b * S_LEN + q0) * (HN * 512) + h * 512;
    for (int r = 0; r < 16; ++r) {
        float inv = 1.f / fac[r];
        for (int c = lane; c < 512; c += 32)
            op[(long)r * (HN * 512) + c] = f2bf(outacc[r * 512 + c] * inv);
    }
}

// ---------- host launcher ----------
extern "C" void kernel_launch(void* const* d_in, const int* in_sizes, int n_in,
                              void* d_out, int out_size, void* d_ws, size_t ws_size,
                              hipStream_t stream) {
    (void)in_sizes; (void)n_in; (void)out_size; (void)ws_size;
    const float* x      = (const float*)d_in[0];
    const float* wq     = (const float*)d_in[1];
    const float* wkva   = (const float*)d_in[2];
    const float* kvnw   = (const float*)d_in[3];
    const float* wkvb   = (const float*)d_in[4];
    const float* wo     = (const float*)d_in[5];
    const float* freqs  = (const float*)d_in[6];
    // d_in[7] mask, d_in[8] start_pos: causality implemented directly.

    const long nX = 2L * 2048 * 2048;        // 8388608
    const long nWq = 3072L * 2048;
    const long nWkva = 576L * 2048;
    const long nWkvb = 4096L * 512;
    const long nWo = 2048L * 2048;
    const long BS = 4096;

    char* w = (char*)d_ws;
    auto carve = [&](long elems) {
        void* p = (void*)w;
        w += ((elems * 2 + 255) & ~255L);
        return (unsigned short*)p;
    };
    unsigned short* xb     = carve(nX);
    unsigned short* wqb    = carve(nWq);
    unsigned short* wkab   = carve(nWkva);
    unsigned short* wkbb   = carve(nWkvb);
    unsigned short* wob    = carve(nWo);
    unsigned short* wkb1t  = carve(16L * 512 * 128);
    unsigned short* qraw   = carve(BS * 3072);
    unsigned short* kvf    = carve(BS * 576);
    unsigned short* qn     = carve(BS * 2048);
    unsigned short* qp     = carve(BS * 1024);
    unsigned short* qabsb  = carve(BS * 16 * 512);
    unsigned short* kvb    = carve(2L * 2048 * 512);
    unsigned short* kpeb   = carve(2L * 2048 * 64);
    unsigned short* vtb    = carve(2L * 512 * 2048);
    unsigned short* attnb  = carve(BS * 16 * 512);
    unsigned short* ofb    = carve(BS * 2048);

    float ms = 0.1f * logf(40.0f) + 1.0f;
    float scale = ms * ms / sqrtf(192.0f);

    // converts
    cvt_f32_bf16<<<(unsigned)((nX + 255) / 256), 256, 0, stream>>>(x, xb, nX);
    cvt_f32_bf16<<<(unsigned)((nWq + 255) / 256), 256, 0, stream>>>(wq, wqb, nWq);
    cvt_f32_bf16<<<(unsigned)((nWkva + 255) / 256), 256, 0, stream>>>(wkva, wkab, nWkva);
    cvt_f32_bf16<<<(unsigned)((nWkvb + 255) / 256), 256, 0, stream>>>(wkvb, wkbb, nWkvb);
    cvt_f32_bf16<<<(unsigned)((nWo + 255) / 256), 256, 0, stream>>>(wo, wob, nWo);

    // wkv_b NOPE block: (h, d=128, c=512) -> (h, c=512, d=128)
    transpose_u16<<<dim3((128 * 512 + 255) / 256, 1, 16), 256, 0, stream>>>(
        wkbb, wkb1t, 128, 512, 256L * 512, 512L * 128);

    // GEMM1: q = x * wq^T  (4096 x 3072, K=2048) -> bf16
    gemm_nt_bf16<<<dim3(3072 / 128, 32, 1), 256, 0, stream>>>(
        xb, wqb, nullptr, qraw, 4096, 3072, 2048, 2048, 2048, 3072, 0, 0, 0);
    // GEMM2: kv_full = x * wkv_a^T (4096 x 576) -> bf16
    gemm_nt_bf16<<<dim3((576 + 127) / 128, 32, 1), 256, 0, stream>>>(
        xb, wkab, nullptr, kvf, 4096, 576, 2048, 2048, 2048, 576, 0, 0, 0);

    rope_q<<<4096, 256, 0, stream>>>(qraw, freqs, qn, qp);
    kv_prep<<<4096, 256, 0, stream>>>(kvf, kvnw, freqs, kvb, kpeb);

    // Vt: (b, t, c) -> (b, c, t)
    transpose_u16<<<dim3((2048 * 512 + 255) / 256, 1, 2), 256, 0, stream>>>(
        kvb, vtb, 2048, 512, 2048L * 512, 512L * 2048);

    // GEMM3 (per head): q_abs = q_nope * wkv_b_nope^T (4096 x 512, K=128)
    gemm_nt_bf16<<<dim3(512 / 128, 32, 16), 256, 0, stream>>>(
        qn, wkb1t, nullptr, qabsb, 4096, 512, 128, 2048, 128, 8192, 128, 512L * 128, 512);

    // attention
    mla_attn<<<dim3(128, 16, 2), 32, 0, stream>>>(qabsb, qp, kvb, kpeb, vtb, attnb, scale);

    // GEMM4 (per head): out2 = attn * wkv_b_v (4096 x 128, K=512)
    gemm_nt_bf16<<<dim3(1, 32, 16), 256, 0, stream>>>(
        attnb, wkbb + 128L * 512, nullptr, ofb, 4096, 128, 512, 8192, 512, 2048, 512, 256L * 512, 128);

    // GEMM5: out = out_flat * wo^T (4096 x 2048, K=2048) -> fp32 d_out
    gemm_nt_bf16<<<dim3(16, 32, 1), 256, 0, stream>>>(
        ofb, wob, (float*)d_out, nullptr, 4096, 2048, 2048, 2048, 2048, 2048, 0, 0, 0);
}
